// SimplePatchTSTEncoder_22600117912083
// MI455X (gfx1250) — matrix operand order repaired and checked by reading, not compiled, not run on hardware
//
#include <hip/hip_runtime.h>
#include <hip/hip_bf16.h>
#include <math.h>

// ---------------- problem constants (match reference) ----------------
#define DMODEL 512
#define NHEAD  16
#define HDIM   32          // DMODEL / NHEAD
#define FFDIM  2048
#define NLAYER 3
#define PLEN   16
#define STRIDE 8
#define CWIN   2048
#define NPATCH 255         // (CWIN - PLEN)/STRIDE + 1
#define NVAR   256
#define MTOK   (NVAR * NPATCH)   // 65280 (divisible by 64)
#define NVCHUNK 32               // attention processed 32 vars at a time
#define SCLD   256               // padded scores leading dim
#define SCSZ   (SCLD * SCLD)     // padded scores per (nv, head)

typedef __attribute__((ext_vector_type(16))) _Float16 v16h;
typedef __attribute__((ext_vector_type(8)))  _Float16 v8h;
typedef __attribute__((ext_vector_type(4)))  _Float16 v4h;
typedef __attribute__((ext_vector_type(8)))  float    v8f;

#define LDSSTR 40   // half-element LDS row stride (80 B: 16B-aligned, bank-skewed)

enum { EPI_NONE = 0, EPI_BIAS = 1, EPI_BIAS_GELU = 2, EPI_BIAS_RES = 3 };

// convert 8 fp32 -> 8 f16 and store as ONE 16B LDS write (ds_store_b128)
__device__ __forceinline__ void cvt_store8(_Float16* dst, float4 a, float4 b) {
    v8h v;
    v[0] = (_Float16)a.x; v[1] = (_Float16)a.y;
    v[2] = (_Float16)a.z; v[3] = (_Float16)a.w;
    v[4] = (_Float16)b.x; v[5] = (_Float16)b.y;
    v[6] = (_Float16)b.z; v[7] = (_Float16)b.w;
    *(v8h*)dst = v;
}

// convert 4 fp32 -> 4 f16 and store as ONE 8B LDS write (ds_store_b64)
__device__ __forceinline__ void cvt_store4(_Float16* dst, float4 a) {
    v4h v;
    v[0] = (_Float16)a.x; v[1] = (_Float16)a.y;
    v[2] = (_Float16)a.z; v[3] = (_Float16)a.w;
    *(v4h*)dst = v;
}

// build v16h fragment from two 16B LDS vector loads (ds_load_b128 x2)
__device__ __forceinline__ v16h frag16(const _Float16* row, int off_lo, int off_hi) {
    v8h lo = *(const v8h*)(row + off_lo);
    v8h hi = *(const v8h*)(row + off_hi);
    v16h f;
#pragma unroll
    for (int i = 0; i < 8; ++i) { f[i] = lo[i]; f[i + 8] = hi[i]; }
    return f;
}

// ---------------------------------------------------------------------
// Token GEMM (guard-free):  C[M,N] = A[M,K] @ W[N,K]^T  (+ epilogue)
// Requires M%64==0, N%64==0, K%32==0.
// Block: 256 threads = 8 waves; tile 64(M) x 64(N); K step 32.
// Waves 4x2; each wave: one A fragment feeds TWO wmma (16x32 of C).
// Double-buffered LDS staging: one barrier per K step.
// ---------------------------------------------------------------------
template <int EPI>
__global__ __launch_bounds__(256) void gemm_wmma_64x64(
    const float* __restrict__ A, const float* __restrict__ W,
    const float* __restrict__ bias, const float* __restrict__ resid,
    float* __restrict__ C, int M, int N, int K, int lda, int ldb, int ldc)
{
    __shared__ __align__(16) _Float16 As[2][64 * LDSSTR];
    __shared__ __align__(16) _Float16 Bs[2][64 * LDSSTR];

    const int  tid = threadIdx.x;
    const long M0  = (long)blockIdx.y * 64;
    const int  N0  = blockIdx.x * 64;

    // staging coords: 64 rows x 32 k, 8 floats per thread
    const int r  = tid >> 2;          // 0..63
    const int kq = (tid & 3) * 8;     // 0,8,16,24
    const float* aptr = A + (M0 + r) * (long)lda + kq;
    const float* bptr = W + (long)(N0 + r) * ldb + kq;
    _Float16* da0 = &As[0][r * LDSSTR + kq];
    _Float16* db0 = &Bs[0][r * LDSSTR + kq];
    _Float16* da1 = &As[1][r * LDSSTR + kq];
    _Float16* db1 = &Bs[1][r * LDSSTR + kq];

    // stage chunk 0 into buffer 0
    cvt_store8(da0, *(const float4*)(aptr), *(const float4*)(aptr + 4));
    cvt_store8(db0, *(const float4*)(bptr), *(const float4*)(bptr + 4));

    const int wave = tid >> 5, lane = tid & 31;
    const int wm = wave & 3;          // M subtile of 16
    const int wn = wave >> 2;         // 0/1 -> N subtile of 32
    const int hsel = lane >> 4, l15 = lane & 15;

    const _Float16* arow0 = &As[0][(wm * 16 + l15) * LDSSTR];
    const _Float16* brow0 = &Bs[0][(wn * 32 + l15) * LDSSTR];
    const _Float16* arow1 = &As[1][(wm * 16 + l15) * LDSSTR];
    const _Float16* brow1 = &Bs[1][(wn * 32 + l15) * LDSSTR];

    v8f acc0 = {}, acc1 = {};
    int buf = 0;
    for (int k0 = 0; k0 < K; k0 += 32) {
        __syncthreads();
        if (k0 + 32 < K) {   // stage next chunk into the other buffer
            const float* ap = aptr + k0 + 32;
            const float* bp = bptr + k0 + 32;
            float4 a0 = *(const float4*)(ap);
            float4 a1 = *(const float4*)(ap + 4);
            float4 b0 = *(const float4*)(bp);
            float4 b1 = *(const float4*)(bp + 4);
            cvt_store8(buf ? da0 : da1, a0, a1);
            cvt_store8(buf ? db0 : db1, b0, b1);
        }
        // compute current chunk
        const _Float16* ar = buf ? arow1 : arow0;
        const _Float16* br = buf ? brow1 : brow0;
        // A 16x32 layout: lanes 0-15 K 0..15; lanes 16-31 K 16..31
        v16h a  = frag16(ar, hsel * 16, hsel * 16 + 8);
        // B 32x16 layout: lane n=l&15; lanes 0-15 K 0..15, lanes 16-31 K 16..31
        v16h b0 = frag16(br, hsel * 16, hsel * 16 + 8);
        v16h b1 = frag16(br + 16 * LDSSTR, hsel * 16, hsel * 16 + 8);
        acc0 = __builtin_amdgcn_wmma_f32_16x16x32_f16(false, a, false, b0,
                                                      (short)0, acc0, false, false);
        acc1 = __builtin_amdgcn_wmma_f32_16x16x32_f16(false, a, false, b1,
                                                      (short)0, acc1, false, false);
        buf ^= 1;
    }

    // epilogue: C layout VGPR rr -> M = rr + 8*hsel, N = l&15
    const int col0 = N0 + wn * 32 + l15;
    const int col1 = col0 + 16;
    float bias0 = 0.0f, bias1 = 0.0f;
    if (EPI != EPI_NONE) { bias0 = bias[col0]; bias1 = bias[col1]; }
#pragma unroll
    for (int rr = 0; rr < 8; ++rr) {
        const long row = M0 + wm * 16 + hsel * 8 + rr;
        float v0 = acc0[rr] + bias0;
        float v1 = acc1[rr] + bias1;
        if (EPI == EPI_BIAS_GELU) {
            v0 = 0.5f * v0 * (1.0f + erff(v0 * 0.70710678118654752f));
            v1 = 0.5f * v1 * (1.0f + erff(v1 * 0.70710678118654752f));
        }
        if (EPI == EPI_BIAS_RES) {
            v0 += resid[row * ldc + col0];
            v1 += resid[row * ldc + col1];
        }
        C[row * ldc + col0] = v0;
        C[row * ldc + col1] = v1;
    }
}

// ---------------------------------------------------------------------
// Attention GEMM (edge variant): 64x32 tile, batched over blockIdx.z.
//   B_NK == true : B stored row-major [N,K]  (Q @ K^T)
//   B_NK == false: B stored row-major [K,N]  (P @ V)
// Staging is UNGUARDED (over-reads stay inside allocated workspace and
// only ever feed epilogue-dropped rows/cols, or are multiplied by the
// zero pad column of the scores buffer). Only stores are guarded.
// ---------------------------------------------------------------------
template <bool B_NK>
__global__ __launch_bounds__(256) void gemm_wmma_edge(
    const float* __restrict__ A, const float* __restrict__ B,
    float* __restrict__ C,
    int M, int N, int K, int lda, int ldb, int ldc,
    int Z0, long sA1, long sA0, long sB1, long sB0, long sC1, long sC0)
{
    __shared__ __align__(16) _Float16 As[64 * LDSSTR];
    __shared__ __align__(16) _Float16 Bs[32 * LDSSTR];

    const int tid = threadIdx.x;
    const int z   = blockIdx.z;
    const int z1  = z / Z0;
    const int z0  = z % Z0;
    const long baseA = (long)z1 * sA1 + (long)z0 * sA0;
    const long baseB = (long)z1 * sB1 + (long)z0 * sB0;
    const long baseC = (long)z1 * sC1 + (long)z0 * sC0;

    const int M0 = blockIdx.y * 64;
    const int N0 = blockIdx.x * 32;

    const int wave = tid >> 5, lane = tid & 31;
    const int wm = wave & 3, wn = wave >> 2;
    const int hsel = lane >> 4, l15 = lane & 15;

    v8f acc = {};

    for (int k0 = 0; k0 < K; k0 += 32) {
        // ---- stage A (64 x 32), unconditional vector loads ----
        {
            const int r  = tid >> 2;
            const int kq = (tid & 3) * 8;
            const float* src = A + baseA + (long)(M0 + r) * lda + k0 + kq;
            cvt_store8(&As[r * LDSSTR + kq],
                       *(const float4*)(src), *(const float4*)(src + 4));
        }
        // ---- stage B into Bs[n][k] (32 x 32) ----
        if (B_NK) {
            const int n  = tid >> 3;
            const int kq = (tid & 7) * 4;
            const float* src = B + baseB + (long)(N0 + n) * ldb + k0 + kq;
            cvt_store4(&Bs[n * LDSSTR + kq], *(const float4*)(src));
        } else {
            const int k  = tid >> 3;
            const int nq = (tid & 7) * 4;
            const float* src = B + baseB + (long)(k0 + k) * ldb + N0 + nq;
            float4 f = *(const float4*)(src);
            Bs[(nq + 0) * LDSSTR + k] = (_Float16)f.x;
            Bs[(nq + 1) * LDSSTR + k] = (_Float16)f.y;
            Bs[(nq + 2) * LDSSTR + k] = (_Float16)f.z;
            Bs[(nq + 3) * LDSSTR + k] = (_Float16)f.w;
        }
        __syncthreads();

        v16h a = frag16(&As[(wm * 16 + l15) * LDSSTR], hsel * 16, hsel * 16 + 8);
        v16h b = frag16(&Bs[(wn * 16 + l15) * LDSSTR], hsel * 16, hsel * 16 + 8);
        acc = __builtin_amdgcn_wmma_f32_16x16x32_f16(false, a, false, b,
                                                     (short)0, acc, false, false);
        __syncthreads();
    }

    const int col = N0 + wn * 16 + l15;
    if (col < N) {
#pragma unroll
        for (int rr = 0; rr < 8; ++rr) {
            const int row = M0 + wm * 16 + hsel * 8 + rr;
            if (row < M)
                C[baseC + (long)row * ldc + col] = acc[rr];
        }
    }
}

// ---------------------------------------------------------------------
// Patch embedding: h[nv,p,d] = sum_j x[nv, 8p+j]*pw[d,j] + pb[d] + pos[p,d]
// ---------------------------------------------------------------------
__global__ __launch_bounds__(256) void patch_embed(
    const float* __restrict__ x, const float* __restrict__ pw,
    const float* __restrict__ pb, const float* __restrict__ pos,
    float* __restrict__ h)
{
    const int bp = blockIdx.x;            // nv*NPATCH + p
    const int nv = bp / NPATCH;
    const int p  = bp % NPATCH;
    __shared__ float xs[PLEN];
    if (threadIdx.x < PLEN)
        xs[threadIdx.x] = x[(long)nv * CWIN + p * STRIDE + threadIdx.x];
    __syncthreads();
#pragma unroll 2
    for (int d = threadIdx.x; d < DMODEL; d += 256) {
        float s = pb[d] + pos[(long)p * DMODEL + d];
#pragma unroll
        for (int j = 0; j < PLEN; ++j) s += xs[j] * pw[d * PLEN + j];
        h[(long)bp * DMODEL + d] = s;
    }
}

// ---------------------------------------------------------------------
// Row softmax with scale over padded scores rows (stride SCLD=256).
// grid = Z * NPATCH rows; pad column 255 is written as 0 (stays zero).
// ---------------------------------------------------------------------
__global__ __launch_bounds__(256) void softmax_rows(float* __restrict__ s,
                                                    float scale)
{
    const int zz = blockIdx.x / NPATCH;
    const int q  = blockIdx.x % NPATCH;
    float* p = s + (long)zz * SCSZ + (long)q * SCLD;
    const int t = threadIdx.x;
    __shared__ float red[256];
    const float v = (t < NPATCH) ? p[t] * scale : -3.402823466e38f;
    red[t] = v; __syncthreads();
#pragma unroll
    for (int o = 128; o > 0; o >>= 1) {
        if (t < o) red[t] = fmaxf(red[t], red[t + o]);
        __syncthreads();
    }
    const float mx = red[0]; __syncthreads();
    const float e = (t < NPATCH) ? __expf(v - mx) : 0.0f;
    red[t] = e; __syncthreads();
#pragma unroll
    for (int o = 128; o > 0; o >>= 1) {
        if (t < o) red[t] += red[t + o];
        __syncthreads();
    }
    p[t] = e * (1.0f / red[0]);   // col 255 -> 0, keeps pad zero
}

// ---------------------------------------------------------------------
// LayerNorm over rows of DMODEL=512 (block 256, 2 elems/thread).
// ---------------------------------------------------------------------
__global__ __launch_bounds__(256) void layernorm_rows(
    const float* __restrict__ in, const float* __restrict__ sc,
    const float* __restrict__ bs, float* __restrict__ out)
{
    const long row = blockIdx.x;
    const float* p = in + row * DMODEL;
    const int t = threadIdx.x;
    const float v0 = p[t], v1 = p[t + 256];
    __shared__ float red[256];
    red[t] = v0 + v1; __syncthreads();
#pragma unroll
    for (int o = 128; o > 0; o >>= 1) {
        if (t < o) red[t] += red[t + o];
        __syncthreads();
    }
    const float mu = red[0] * (1.0f / DMODEL); __syncthreads();
    const float d0 = v0 - mu, d1 = v1 - mu;
    red[t] = d0 * d0 + d1 * d1; __syncthreads();
#pragma unroll
    for (int o = 128; o > 0; o >>= 1) {
        if (t < o) red[t] += red[t + o];
        __syncthreads();
    }
    const float r = rsqrtf(red[0] * (1.0f / DMODEL) + 1e-5f);
    out[row * DMODEL + t]       = d0 * r * sc[t]       + bs[t];
    out[row * DMODEL + t + 256] = d1 * r * sc[t + 256] + bs[t + 256];
}

// ---------------------------------------------------------------------
extern "C" void kernel_launch(void* const* d_in, const int* in_sizes, int n_in,
                              void* d_out, int out_size, void* d_ws, size_t ws_size,
                              hipStream_t stream)
{
    const float* x       = (const float*)d_in[0];
    const float* patch_w = (const float*)d_in[1];
    const float* patch_b = (const float*)d_in[2];
    const float* pos     = (const float*)d_in[3];
    const float* qkv_w   = (const float*)d_in[4];
    const float* qkv_b   = (const float*)d_in[5];
    const float* out_w   = (const float*)d_in[6];
    const float* out_b   = (const float*)d_in[7];
    const float* lin1_w  = (const float*)d_in[8];
    const float* lin1_b  = (const float*)d_in[9];
    const float* lin2_w  = (const float*)d_in[10];
    const float* lin2_b  = (const float*)d_in[11];
    const float* ln1_s   = (const float*)d_in[12];
    const float* ln1_b   = (const float*)d_in[13];
    const float* ln2_s   = (const float*)d_in[14];
    const float* ln2_b   = (const float*)d_in[15];
    const float* lnf_s   = (const float*)d_in[16];
    const float* lnf_b   = (const float*)d_in[17];
    float* out = (float*)d_out;

    // ---- carve workspace ----
    float* ws  = (float*)d_ws;
    float* h   = ws;  ws += (long)MTOK * DMODEL;   // activations
    float* tmp = ws;  ws += (long)MTOK * DMODEL;   // pre-LN residual sums
    float* o   = ws;  ws += (long)MTOK * DMODEL;   // attention output
    float* un  = ws;  ws += (long)MTOK * FFDIM;    // union: qkv (M*1536) / ff (M*2048)
    float* sc  = ws;                               // NVCHUNK*H*256*256 padded scores

    const dim3 blk(256);
    const float scale = 0.17677669529663687f;      // 1/sqrt(HDIM)
    const int D3 = 3 * DMODEL;

    // zero the padded scores region once: pad row/col stay 0 forever after
    hipMemsetAsync(sc, 0, (size_t)NVCHUNK * NHEAD * SCSZ * sizeof(float), stream);

    patch_embed<<<MTOK, blk, 0, stream>>>(x, patch_w, patch_b, pos, h);

    for (int l = 0; l < NLAYER; ++l) {
        // qkv = h @ qkv_w[l].T + qkv_b[l]        -> un [M, 1536]
        gemm_wmma_64x64<EPI_BIAS><<<dim3(D3 / 64, MTOK / 64, 1), blk, 0, stream>>>(
            h, qkv_w + (long)l * D3 * DMODEL, qkv_b + (long)l * D3, nullptr, un,
            MTOK, D3, DMODEL, DMODEL, DMODEL, D3);

        // attention in chunks of NVCHUNK variables
        for (int c = 0; c < NVAR; c += NVCHUNK) {
            const float* qb = un + (long)c * NPATCH * D3;
            // scores[z] = Q K^T  (z = nvLocal*NHEAD + head), into padded buffer
            gemm_wmma_edge<true><<<dim3(8, 4, NVCHUNK * NHEAD), blk, 0, stream>>>(
                qb, qb + DMODEL, sc,
                NPATCH, NPATCH, HDIM, D3, D3, SCLD,
                NHEAD, (long)NPATCH * D3, HDIM,
                       (long)NPATCH * D3, HDIM,
                       (long)NHEAD * SCSZ, (long)SCSZ);
            // softmax(scale * scores)
            softmax_rows<<<NVCHUNK * NHEAD * NPATCH, blk, 0, stream>>>(sc, scale);
            // o = P @ V  (K=256: pad col of P is zero, so no K-edge)
            gemm_wmma_edge<false><<<dim3(1, 4, NVCHUNK * NHEAD), blk, 0, stream>>>(
                sc, qb + 2 * DMODEL, o + (long)c * NPATCH * DMODEL,
                NPATCH, HDIM, SCLD, SCLD, D3, DMODEL,
                NHEAD, (long)NHEAD * SCSZ, (long)SCSZ,
                       (long)NPATCH * D3, HDIM,
                       (long)NPATCH * DMODEL, HDIM);
        }

        // tmp = h + (o @ out_w[l].T + out_b[l]);  h = LN1(tmp)
        gemm_wmma_64x64<EPI_BIAS_RES><<<dim3(DMODEL / 64, MTOK / 64, 1), blk, 0, stream>>>(
            o, out_w + (long)l * DMODEL * DMODEL, out_b + (long)l * DMODEL, h, tmp,
            MTOK, DMODEL, DMODEL, DMODEL, DMODEL, DMODEL);
        layernorm_rows<<<MTOK, blk, 0, stream>>>(tmp, ln1_s + (long)l * DMODEL,
                                                 ln1_b + (long)l * DMODEL, h);

        // f = gelu(h @ lin1_w[l].T + lin1_b[l])  -> un [M, 2048]
        gemm_wmma_64x64<EPI_BIAS_GELU><<<dim3(FFDIM / 64, MTOK / 64, 1), blk, 0, stream>>>(
            h, lin1_w + (long)l * FFDIM * DMODEL, lin1_b + (long)l * FFDIM, nullptr, un,
            MTOK, FFDIM, DMODEL, DMODEL, DMODEL, FFDIM);

        // tmp = h + (f @ lin2_w[l].T + lin2_b[l]);  h = LN2(tmp)
        gemm_wmma_64x64<EPI_BIAS_RES><<<dim3(DMODEL / 64, MTOK / 64, 1), blk, 0, stream>>>(
            un, lin2_w + (long)l * DMODEL * FFDIM, lin2_b + (long)l * DMODEL, h, tmp,
            MTOK, DMODEL, FFDIM, FFDIM, FFDIM, DMODEL);
        layernorm_rows<<<MTOK, blk, 0, stream>>>(tmp, ln2_s + (long)l * DMODEL,
                                                 ln2_b + (long)l * DMODEL, h);
    }

    // final LN -> output
    layernorm_rows<<<MTOK, blk, 0, stream>>>(h, lnf_s, lnf_b, out);
}
